// AMRCorefModel_65463891526241
// MI455X (gfx1250) — matrix operand order repaired and checked by reading, not compiled, not run on hardware
//
#include <hip/hip_runtime.h>
#include <math.h>

#define M_MENT 1024
#define C_ANT  128
#define D_EMB  512
#define FF_DIM 1024
#define NB     3072   // packed cols: [W0a | W0b | ms_W0]

typedef __attribute__((ext_vector_type(16))) __bf16 v16bf;
typedef __attribute__((ext_vector_type(8)))  __bf16 v8bf;
typedef __attribute__((ext_vector_type(8)))  float  v8f;

static __device__ __forceinline__ unsigned short f2bf(float f) {
  unsigned int u = __float_as_uint(f);
  u += 0x7FFFu + ((u >> 16) & 1u);   // round-to-nearest-even
  return (unsigned short)(u >> 16);
}

static __device__ __forceinline__ v16bf cat16(v8bf lo, v8bf hi) {
  return __builtin_shufflevector(lo, hi, 0,1,2,3,4,5,6,7,8,9,10,11,12,13,14,15);
}

static __device__ __forceinline__ v8f wmma_bf16(v16bf a, v16bf b, v8f c) {
  return __builtin_amdgcn_wmma_f32_16x16x32_bf16(false, a, false, b, (short)0, c, false, false);
}

// ---------------- pack kernels ----------------

// Bt[n][k] (bf16, [3072][512]) = transposed concat of sp_W0[0:512], sp_W0[512:1024], ms_W0
__global__ void pack_bt_kernel(const float* __restrict__ spW0,
                               const float* __restrict__ msW0,
                               unsigned short* __restrict__ Bt) {
  int idx = blockIdx.x * 256 + threadIdx.x;
  if (idx >= NB * D_EMB) return;
  int n = idx >> 9, k = idx & 511;
  float v;
  if (n < 1024)       v = spW0[k * FF_DIM + n];
  else if (n < 2048)  v = spW0[(512 + k) * FF_DIM + (n - 1024)];
  else                v = msW0[k * FF_DIM + (n - 2048)];
  Bt[idx] = f2bf(v);
}

// Wct[n][k] (bf16, [1024][512]) = transpose of W0c = sp_W0[1024:1536]
__global__ void pack_wct_kernel(const float* __restrict__ spW0,
                                unsigned short* __restrict__ Wct) {
  int idx = blockIdx.x * 256 + threadIdx.x;
  if (idx >= FF_DIM * D_EMB) return;
  int n = idx >> 9, k = idx & 511;
  Wct[idx] = f2bf(spW0[(1024 + k) * FF_DIM + n]);
}

__global__ void pack_me_kernel(const float* __restrict__ me,
                               unsigned short* __restrict__ meb) {
  int idx = blockIdx.x * 256 + threadIdx.x;
  if (idx >= M_MENT * D_EMB) return;
  meb[idx] = f2bf(me[idx]);
}

// SB[s*10+b][n] = speaker_emb[s]@W0d + bucket_emb[b]@W0e + sp_b0  (20 x 1024)
__global__ void sb_kernel(const float* __restrict__ spW0,
                          const float* __restrict__ spb0,
                          const float* __restrict__ spe,
                          const float* __restrict__ bke,
                          float* __restrict__ SB) {
  int idx = blockIdx.x * 256 + threadIdx.x;
  if (idx >= 20 * FF_DIM) return;
  int row = idx >> 10, n = idx & 1023;
  int s = row / 10, b = row % 10;
  float acc = spb0[n];
  #pragma unroll
  for (int j = 0; j < 20; ++j) acc += spe[s * 20 + j] * spW0[(1536 + j) * FF_DIM + n];
  #pragma unroll
  for (int j = 0; j < 20; ++j) acc += bke[b * 20 + j] * spW0[(1556 + j) * FF_DIM + n];
  SB[idx] = acc;
}

// ---------------- P = me_bf16 @ Bt  (WMMA), P fp32 [1024][3072] ----------------

__global__ __launch_bounds__(256) void gemmP_kernel(
    const unsigned short* __restrict__ A,   // [1024][512] bf16
    const unsigned short* __restrict__ Bt,  // [3072][512] bf16 (N-major)
    float* __restrict__ P) {
  int wid = blockIdx.x * 8 + (threadIdx.x >> 5);      // 12288 tiles = 64 x 192
  int rt = wid / (NB / 16), ct = wid % (NB / 16);
  int lane = threadIdx.x & 31, lr = lane & 15, lh = lane >> 4;
  const unsigned short* abase = A  + (rt * 16 + lr) * D_EMB;
  const unsigned short* bbase = Bt + (ct * 16 + lr) * D_EMB;
  v8f acc = {0.f,0.f,0.f,0.f,0.f,0.f,0.f,0.f};
  #pragma unroll
  for (int kc = 0; kc < 16; ++kc) {
    const unsigned short* ap = abase + kc * 32 + lh * 8;
    v16bf a = cat16(*(const v8bf*)ap, *(const v8bf*)(ap + 16));
    const unsigned short* bp = bbase + kc * 32 + lh * 8;
    v16bf b = cat16(*(const v8bf*)bp, *(const v8bf*)(bp + 16));
    acc = wmma_bf16(a, b, acc);
  }
  #pragma unroll
  for (int v = 0; v < 8; ++v) {
    int row = rt * 16 + v + 8 * lh;
    int col = ct * 16 + lr;
    P[row * NB + col] = acc[v];
  }
}

// ---------------- mention scores: relu(P[:,2048:]+b0) @ ms_W1 + ms_b1 ----------------

__global__ __launch_bounds__(256) void msscore_kernel(
    const float* __restrict__ P, const float* __restrict__ b0,
    const float* __restrict__ W1, const float* __restrict__ b1,
    float* __restrict__ out) {
  int m = blockIdx.x * 8 + (threadIdx.x >> 5);
  int lane = threadIdx.x & 31;
  float acc = 0.f;
  for (int c = lane; c < FF_DIM; c += 32) {
    float h = fmaxf(P[m * NB + 2048 + c] + b0[c], 0.f);
    acc = fmaf(h, W1[c], acc);
  }
  #pragma unroll
  for (int off = 16; off >= 1; off >>= 1) acc += __shfl_xor(acc, off, 32);
  if (lane == 0) out[m] = acc + b1[0];
}

// ---------------- pair scorer: the 137-GFLOP kernel ----------------
// block = one mention m, 256 thr = 8 waves, wave w owns c-tile [16w,16w+16)
// LDS: q[c][k] bf16, row stride 520 elems (+16B pad -> conflict-free b128 loads)

__global__ __launch_bounds__(256) void pairscore_kernel(
    const float* __restrict__ me,            // [1024][512] fp32
    const unsigned short* __restrict__ Wct,  // [1024][512] bf16
    const float* __restrict__ P,             // [1024][3072]
    const float* __restrict__ SB,            // [20][1024]
    const float* __restrict__ w1,            // sp_W1 [1024]
    const float* __restrict__ b1,            // sp_b1 [1]
    const float* __restrict__ msc,           // mention_scores [1024]
    const int* __restrict__ spk,             // [1024]
    float* __restrict__ ant_scores) {        // [1024][128]
  extern __shared__ char smem[];
  unsigned short* qlds = (unsigned short*)smem;      // [128][520]
  const int m = blockIdx.x;
  const int tid = threadIdx.x;

  // stage q = src (.) ant, fp32 products -> bf16 in LDS
  for (int u = tid; u < C_ANT * (D_EMB / 4); u += 256) {
    int c = u >> 7;
    int k0 = (u & 127) * 4;
    int a = m - 1 - c; if (a < 0) a = 0;
    const float4 s4 = *(const float4*)(me + m * D_EMB + k0);
    const float4 a4 = *(const float4*)(me + a * D_EMB + k0);
    unsigned int p0 = (unsigned int)f2bf(s4.x * a4.x) | ((unsigned int)f2bf(s4.y * a4.y) << 16);
    unsigned int p1 = (unsigned int)f2bf(s4.z * a4.z) | ((unsigned int)f2bf(s4.w * a4.w) << 16);
    *(uint2*)(qlds + c * 520 + k0) = make_uint2(p0, p1);
  }
  __syncthreads();

  const int wid = tid >> 5;
  const int lane = tid & 31, lr = lane & 15, lh = lane >> 4;

  // cache all 16 A-fragments for this wave's 16-row tile in VGPRs
  v16bf afrag[16];
  #pragma unroll
  for (int kc = 0; kc < 16; ++kc) {
    const unsigned short* ap = qlds + (wid * 16 + lr) * 520 + kc * 32 + lh * 8;
    afrag[kc] = cat16(*(const v8bf*)ap, *(const v8bf*)(ap + 16));
  }

  // per-lane row metadata for the 8 D-rows this lane produces (rows v + 8*lh)
  const int spkm = spk[m];
  int pbo[8]; int sbo[8]; int aidx[8]; float partial[8];
  #pragma unroll
  for (int v = 0; v < 8; ++v) {
    int c = wid * 16 + v + 8 * lh;
    int a = m - 1 - c; if (a < 0) a = 0;
    aidx[v] = a;
    pbo[v] = a * NB + 1024;
    int d = c + 1;
    int b = (d <= 4) ? d : (31 - __clz(d) + 3);
    if (b > 9) b = 9;
    int s = (spk[a] == spkm) ? 1 : 0;
    sbo[v] = (s * 10 + b) * FF_DIM;
    partial[v] = 0.f;
  }

  const float* Pm = P + m * NB;
  for (int nt = 0; nt < FF_DIM / 16; ++nt) {
    int col = nt * 16 + lr;
    v8f acc = {0.f,0.f,0.f,0.f,0.f,0.f,0.f,0.f};
    const unsigned short* bbase = Wct + col * D_EMB;
    #pragma unroll
    for (int kc = 0; kc < 16; ++kc) {
      const unsigned short* bp = bbase + kc * 32 + lh * 8;
      v16bf b = cat16(*(const v8bf*)bp, *(const v8bf*)(bp + 16));
      acc = wmma_bf16(afrag[kc], b, acc);
    }
    float pa = Pm[col];
    float w  = w1[col];
    #pragma unroll
    for (int v = 0; v < 8; ++v) {
      float h = acc[v] + pa + P[pbo[v] + col] + SB[sbo[v] + col];
      h = fmaxf(h, 0.f);
      partial[v] = fmaf(h, w, partial[v]);
    }
  }

  // reduce each row's partial across its 16-lane group
  #pragma unroll
  for (int v = 0; v < 8; ++v) {
    float p = partial[v];
    #pragma unroll
    for (int off = 8; off >= 1; off >>= 1) p += __shfl_xor(p, off, 16);
    partial[v] = p;
  }
  if (lr == 0) {
    const float msm = msc[m];
    const float bb1 = b1[0];
    #pragma unroll
    for (int v = 0; v < 8; ++v) {
      int c = wid * 16 + v + 8 * lh;
      bool mask = (m - 1 - c) >= 0;
      float sc = mask ? (partial[v] + bb1 + msm + msc[aidx[v]]) : -INFINITY;
      ant_scores[m * C_ANT + c] = sc;
    }
  }
}

// ---------------- softmax + clip + renorm + labels + per-m loss ----------------

__global__ __launch_bounds__(128) void softmax_loss_kernel(
    const float* __restrict__ ant, const int* __restrict__ clus,
    float* __restrict__ out, float* __restrict__ lossm) {
  __shared__ float red[128];
  __shared__ int redi[128];
  const int m = blockIdx.x, t = threadIdx.x;
  const float s = ant[m * C_ANT + t];

  red[t] = s; __syncthreads();
  for (int off = 64; off >= 1; off >>= 1) { if (t < off) red[t] = fmaxf(red[t], red[t + off]); __syncthreads(); }
  const float mx = fmaxf(red[0], 0.f);
  __syncthreads();

  const float e = expf(s - mx);
  const float edum = expf(-mx);
  red[t] = e; __syncthreads();
  for (int off = 64; off >= 1; off >>= 1) { if (t < off) red[t] += red[t + off]; __syncthreads(); }
  const float inv = 1.f / (red[0] + edum);
  __syncthreads();

  const float eps = 1e-6f;
  float d    = fminf(fmaxf(e * inv, eps), 1.f - eps);
  float ddum = fminf(fmaxf(edum * inv, eps), 1.f - eps);
  red[t] = d; __syncthreads();
  for (int off = 64; off >= 1; off >>= 1) { if (t < off) red[t] += red[t + off]; __syncthreads(); }
  const float inv2 = 1.f / (red[0] + ddum);
  __syncthreads();

  const float fd = d * inv2, fdum = ddum * inv2;
  out[m * (C_ANT + 1) + 1 + t] = fd;
  if (t == 0) out[m * (C_ANT + 1)] = fdum;

  const int cm = clus[m];
  int a = m - 1 - t; const bool mask = a >= 0; if (a < 0) a = 0;
  const int ac = mask ? clus[a] : 0;
  const bool lab = (ac == cm) && (cm > 0);
  redi[t] = lab ? 1 : 0; __syncthreads();
  for (int off = 64; off >= 1; off >>= 1) { if (t < off) redi[t] += redi[t + off]; __syncthreads(); }
  const int any = redi[0];
  red[t] = lab ? -logf(fd) : 0.f;
  __syncthreads();
  for (int off = 64; off >= 1; off >>= 1) { if (t < off) red[t] += red[t + off]; __syncthreads(); }
  if (t == 0) lossm[m] = red[0] + (any ? 0.f : -logf(fdum));
}

__global__ __launch_bounds__(256) void loss_reduce_kernel(
    const float* __restrict__ lossm, float* __restrict__ out) {
  __shared__ float red[256];
  float acc = 0.f;
  for (int i = threadIdx.x; i < M_MENT; i += 256) acc += lossm[i];
  red[threadIdx.x] = acc; __syncthreads();
  for (int off = 128; off >= 1; off >>= 1) { if ((int)threadIdx.x < off) red[threadIdx.x] += red[threadIdx.x + off]; __syncthreads(); }
  if (threadIdx.x == 0) out[M_MENT * (C_ANT + 1)] = red[0];   // B==1 -> mean == sum
}

// ---------------- launch ----------------

extern "C" void kernel_launch(void* const* d_in, const int* in_sizes, int n_in,
                              void* d_out, int out_size, void* d_ws, size_t ws_size,
                              hipStream_t stream) {
  (void)in_sizes; (void)n_in; (void)out_size; (void)ws_size;
  const float* me    = (const float*)d_in[0];
  const int*   clus  = (const int*)d_in[1];
  const int*   spk   = (const int*)d_in[2];
  const float* msW0  = (const float*)d_in[3];
  const float* msb0  = (const float*)d_in[4];
  const float* msW1  = (const float*)d_in[5];
  const float* msb1  = (const float*)d_in[6];
  const float* spW0  = (const float*)d_in[7];
  const float* spb0  = (const float*)d_in[8];
  const float* spW1  = (const float*)d_in[9];
  const float* spb1  = (const float*)d_in[10];
  const float* spe   = (const float*)d_in[11];
  const float* bke   = (const float*)d_in[12];

  char* ws = (char*)d_ws;
  unsigned short* Bt  = (unsigned short*)(ws + 0);          //  3 MB
  unsigned short* Wct = (unsigned short*)(ws + 3145728);    //  1 MB
  unsigned short* Meb = (unsigned short*)(ws + 4194304);    //  1 MB
  float* P    = (float*)(ws + 5242880);                     // 12 MB
  float* SB   = (float*)(ws + 17825792);                    // 80 KB
  float* Msc  = (float*)(ws + 17907712);                    //  4 KB
  float* Ant  = (float*)(ws + 17911808);                    // 512 KB
  float* Lm   = (float*)(ws + 18436096);                    //  4 KB
  float* out  = (float*)d_out;

  pack_bt_kernel <<<6144, 256, 0, stream>>>(spW0, msW0, Bt);
  pack_wct_kernel<<<2048, 256, 0, stream>>>(spW0, Wct);
  pack_me_kernel <<<2048, 256, 0, stream>>>(me, Meb);
  sb_kernel      <<<80,   256, 0, stream>>>(spW0, spb0, spe, bke, SB);
  gemmP_kernel   <<<1536, 256, 0, stream>>>(Meb, Bt, P);
  msscore_kernel <<<128,  256, 0, stream>>>(P, msb0, msW1, msb1, Msc);

  const size_t lds_bytes = 128 * 520 * sizeof(unsigned short);   // 133120 B
  hipFuncSetAttribute((const void*)pairscore_kernel,
                      hipFuncAttributeMaxDynamicSharedMemorySize, (int)lds_bytes);
  pairscore_kernel<<<1024, 256, lds_bytes, stream>>>(me, Wct, P, SB, spW1, spb1, Msc, spk, Ant);

  softmax_loss_kernel<<<1024, 128, 0, stream>>>(Ant, clus, out, Lm);
  loss_reduce_kernel <<<1,    256, 0, stream>>>(Lm, out);
}
